// DgaWinSequence_61186104099022
// MI455X (gfx1250) — matrix operand order repaired and checked by reading, not compiled, not run on hardware
//
#include <hip/hip_runtime.h>
#include <hip/hip_bf16.h>

typedef __attribute__((ext_vector_type(16))) _Float16 v16h;
typedef __attribute__((ext_vector_type(8)))  _Float16 v8h;
typedef __attribute__((ext_vector_type(8)))  float    v8f;

#define B_  16
#define T_  256
#define IN_ 6
#define HID_ 256
#define FEAT_ 64
#define STATE_ 64
#define MOTOR_ 16
#define ROWS_ (B_*T_)      // 4096
#define UNFOLDS_ 6
#define EPS_ 1e-8f

__device__ __forceinline__ float sigmoid_f(float z) {
    return 1.0f / (1.0f + __expf(-z));
}

// ---------------- Phase 1: H = tanh(x @ pw1 + pb1), stored f16 [4096,256] ----
__global__ void ltc_mlp1(const float* __restrict__ x, const float* __restrict__ pw1,
                         const float* __restrict__ pb1, _Float16* __restrict__ H) {
    int idx  = blockIdx.x * blockDim.x + threadIdx.x;   // 4096*256 threads
    int hcol = idx & (HID_ - 1);
    int row  = idx >> 8;
    const float* xr = x + row * IN_;
    float acc = pb1[hcol];
#pragma unroll
    for (int i = 0; i < IN_; ++i) acc += xr[i] * pw1[i * HID_ + hcol];
    H[idx] = (_Float16)tanhf(acc);
}

// ------- Pack pw2 [256,64] f32 -> WMMA B-fragment layout, contiguous k, f16 ----------
// Bp[((nt*8 + kc)*32 + lane)*16 + h] = pw2[(kc*32 + h + 16*(lane>>4))*64 + nt*16 + (lane&15)]
__global__ void ltc_packB(const float* __restrict__ pw2, _Float16* __restrict__ Bp) {
    int idx  = blockIdx.x * blockDim.x + threadIdx.x;   // 16384 threads
    int h    = idx & 15;
    int lane = (idx >> 4) & 31;
    int kc   = (idx >> 9) & 7;
    int nt   = idx >> 12;
    int K = kc * 32 + h + 16 * (lane >> 4);
    int N = nt * 16 + (lane & 15);
    Bp[idx] = (_Float16)pw2[K * FEAT_ + N];
}

// ---- Phase 2: feats = (H @ pw2 + pb2) * input_w + input_b  via WMMA ---------
// One wave (32 threads) per 16x16 output tile. M tiles = 256, N tiles = 4.
__global__ void ltc_gemm_feats(const _Float16* __restrict__ H, const _Float16* __restrict__ Bp,
                               const float* __restrict__ pb2,
                               const float* __restrict__ in_w, const float* __restrict__ in_b,
                               float* __restrict__ feats) {
    int wave = blockIdx.x;          // 0..1023
    int mt   = wave >> 2;
    int nt   = wave & 3;
    int lane = threadIdx.x;         // 0..31 (wave32)
    int m    = lane & 15;
    int half = lane >> 4;
    int row  = mt * 16 + m;
    int ncol = nt * 16 + (lane & 15);

    float bias = pb2[ncol];
    v8f c;
#pragma unroll
    for (int r = 0; r < 8; ++r) c[r] = bias;

#pragma unroll
    for (int kc = 0; kc < 8; ++kc) {
        int kbase = kc * 32;
        union { v16h v; v8h h2[2]; } ua;
        // A fragment: halfs 0..7 -> K = kbase + 16*half + (0..7)
        //             halfs 8..15 -> K = kbase + 16*half + 8 + (0..7)
        ua.h2[0] = *(const v8h*)(H + row * HID_ + kbase + 16 * half);
        ua.h2[1] = *(const v8h*)(H + row * HID_ + kbase + 16 * half + 8);
        v16h b = *(const v16h*)(Bp + (((nt * 8) + kc) * 32 + lane) * 16);
        c = __builtin_amdgcn_wmma_f32_16x16x32_f16(false, ua.v, false, b,
                                                   (short)0, c, false, false);
    }

    float iw = in_w[ncol], ib = in_b[ncol];
#pragma unroll
    for (int r = 0; r < 8; ++r) {
        int orow = mt * 16 + r + 8 * half;      // D layout: M = r + 8*half
        feats[orow * FEAT_ + ncol] = c[r] * iw + ib;
    }
}

// ---- Phase 3: sensory synapse reduction -> num_s, den_s [4096,64] -----------
__global__ void ltc_sensory(const float* __restrict__ feats,
                            const float* __restrict__ sw,  const float* __restrict__ smu,
                            const float* __restrict__ ssg, const float* __restrict__ ser,
                            float* __restrict__ nums, float* __restrict__ dens) {
    __shared__ float lw[FEAT_*STATE_], lmu[FEAT_*STATE_], lsg[FEAT_*STATE_], ler[FEAT_*STATE_];
    int tid = threadIdx.x;                  // 256
    for (int i = tid; i < FEAT_*STATE_; i += 256) {
        lw[i] = sw[i]; lmu[i] = smu[i]; lsg[i] = ssg[i]; ler[i] = ser[i];
    }
    __syncthreads();
    int j    = tid & 63;                    // post state
    int rsub = tid >> 6;                    // 4 rows per block
    int row  = blockIdx.x * 4 + rsub;
    const float* f = feats + row * FEAT_;
    float sn = 0.f, sd = 0.f;
#pragma unroll 4
    for (int k = 0; k < FEAT_; ++k) {
        float z = (f[k] - lmu[k * STATE_ + j]) * lsg[k * STATE_ + j];
        float a = lw[k * STATE_ + j] * sigmoid_f(z);
        sn += a * ler[k * STATE_ + j];
        sd += a;
    }
    nums[row * STATE_ + j] = sn;
    dens[row * STATE_ + j] = sd;
}

// ---- Phase 4: sequential LTC scan. One block per batch. ---------------------
// 256 threads = 64 post-states x 4 pre-chunks of 16.
__global__ void ltc_scan(const float* __restrict__ nums, const float* __restrict__ dens,
                         const float* __restrict__ w,   const float* __restrict__ mu,
                         const float* __restrict__ sg,  const float* __restrict__ er,
                         const float* __restrict__ gleak, const float* __restrict__ vleak,
                         const float* __restrict__ cm,
                         const float* __restrict__ out_w, const float* __restrict__ out_b,
                         float* __restrict__ out) {
    __shared__ float lw[STATE_*STATE_], lmu[STATE_*STATE_], lsg[STATE_*STATE_], ler[STATE_*STATE_];
    __shared__ float v[STATE_];
    __shared__ float psn[256], psd[256];

    int tid = threadIdx.x;
    int j   = tid & 63;      // post state
    int c   = tid >> 6;      // pre chunk 0..3
    for (int i = tid; i < STATE_*STATE_; i += 256) {
        lw[i] = w[i]; lmu[i] = mu[i]; lsg[i] = sg[i]; ler[i] = er[i];
    }
    if (tid < STATE_) v[tid] = 0.f;

    int b = blockIdx.x;
    float gl = gleak[j], glvl = gleak[j] * vleak[j], cmt = cm[j] * (float)UNFOLDS_;
    float ow = 0.f, ob = 0.f;
    if (j < MOTOR_) { ow = out_w[j]; ob = out_b[j]; }
    __syncthreads();

    for (int t = 0; t < T_; ++t) {
        int row = b * T_ + t;
        float ns = 0.f, ds = 0.f;
        if (c == 0) { ns = nums[row * STATE_ + j]; ds = dens[row * STATE_ + j]; }
        float vnew = 0.f;
        for (int u = 0; u < UNFOLDS_; ++u) {
            float sn = 0.f, sd = 0.f;
            int p0 = c * 16;
#pragma unroll
            for (int pp = 0; pp < 16; ++pp) {
                int p = p0 + pp;
                float z = (v[p] - lmu[p * STATE_ + j]) * lsg[p * STATE_ + j];
                float a = lw[p * STATE_ + j] * sigmoid_f(z);
                sn += a * ler[p * STATE_ + j];
                sd += a;
            }
            psn[tid] = sn; psd[tid] = sd;
            __syncthreads();
            if (c == 0) {
                float SN = psn[j] + psn[64 + j] + psn[128 + j] + psn[192 + j];
                float SD = psd[j] + psd[64 + j] + psd[128 + j] + psd[192 + j];
                float numer = cmt * v[j] + glvl + SN + ns;
                float denom = cmt + gl + SD + ds;
                vnew = numer / (denom + EPS_);
                v[j] = vnew;
            }
            __syncthreads();
        }
        if (c == 0 && j < MOTOR_) out[row * MOTOR_ + j] = vnew * ow + ob;
    }
}

extern "C" void kernel_launch(void* const* d_in, const int* in_sizes, int n_in,
                              void* d_out, int out_size, void* d_ws, size_t ws_size,
                              hipStream_t stream) {
    const float* x        = (const float*)d_in[0];
    const float* pw1      = (const float*)d_in[1];
    const float* pb1      = (const float*)d_in[2];
    const float* pw2      = (const float*)d_in[3];
    const float* pb2      = (const float*)d_in[4];
    const float* input_w  = (const float*)d_in[5];
    const float* input_b  = (const float*)d_in[6];
    const float* s_w      = (const float*)d_in[7];
    const float* s_mu     = (const float*)d_in[8];
    const float* s_sigma  = (const float*)d_in[9];
    const float* s_erev   = (const float*)d_in[10];
    const float* w        = (const float*)d_in[11];
    const float* mu       = (const float*)d_in[12];
    const float* sigma    = (const float*)d_in[13];
    const float* erev     = (const float*)d_in[14];
    const float* gleak    = (const float*)d_in[15];
    const float* vleak    = (const float*)d_in[16];
    const float* cm       = (const float*)d_in[17];
    const float* output_w = (const float*)d_in[18];
    const float* output_b = (const float*)d_in[19];
    float* out = (float*)d_out;

    // Workspace carving (hipMalloc base is 256B aligned).
    char* ws = (char*)d_ws;
    _Float16* H     = (_Float16*)(ws);                          // 4096*256 f16 = 2 MiB
    _Float16* Bp    = (_Float16*)(ws + 2097152);                // 16384 f16   = 32 KiB
    float*    feats = (float*)   (ws + 2097152 + 32768);        // 4096*64 f32 = 1 MiB
    float*    nums  = (float*)   (ws + 2097152 + 32768 + 1048576);
    float*    dens  = (float*)   (ws + 2097152 + 32768 + 2*1048576);

    ltc_mlp1<<<ROWS_, HID_, 0, stream>>>(x, pw1, pb1, H);
    ltc_packB<<<64, 256, 0, stream>>>(pw2, Bp);
    ltc_gemm_feats<<<(ROWS_/16) * (FEAT_/16), 32, 0, stream>>>(H, Bp, pb2, input_w, input_b, feats);
    ltc_sensory<<<ROWS_/4, 256, 0, stream>>>(feats, s_w, s_mu, s_sigma, s_erev, nums, dens);
    ltc_scan<<<B_, 256, 0, stream>>>(nums, dens, w, mu, sigma, erev,
                                     gleak, vleak, cm, output_w, output_b, out);
}
